// LinearCRF_39195871543868
// MI455X (gfx1250) — matrix operand order repaired and checked by reading, not compiled, not run on hardware
//
#include <hip/hip_runtime.h>
#include <math.h>

typedef _Float16 v16h __attribute__((ext_vector_type(16)));
typedef _Float16 v8h  __attribute__((ext_vector_type(8)));
typedef float    v8f  __attribute__((ext_vector_type(8)));

#define NB     32
#define NS     512
#define NL     64
#define TSTART 61
#define TSTOP  62
#define BPW    16      // batches (M rows) per workgroup
#define LN2    0.6931471805599453f
#define INVLN2 1.4426950408889634f

// 8-lane-group max via single-instruction DPP8 max ops (VALU only, no LDS pipe):
// v_max_num_f32_dpp D, S, S dpp8:[...]  =>  D = max(perm8(S), S)
__device__ __forceinline__ float group8_max(float x) {
    float y, z;
    asm("v_max_num_f32_dpp %0, %1, %1 dpp8:[1,0,3,2,5,4,7,6]" : "=v"(y) : "v"(x));
    asm("v_max_num_f32_dpp %0, %1, %1 dpp8:[2,3,0,1,6,7,4,5]" : "=v"(z) : "v"(y));
    asm("v_max_num_f32_dpp %0, %1, %1 dpp8:[4,5,6,7,0,1,2,3]" : "=v"(y) : "v"(z));
    return y;
}

__global__ __launch_bounds__(128)
void crf_fwd_kernel(const float* __restrict__ enc,
                    const float* __restrict__ trans,
                    const int*   __restrict__ lens,
                    const int*   __restrict__ tags,
                    float*       __restrict__ out)
{
    // State: exp(alpha_t[b][j]) = u[b][j] * 2^Cb   with u in [0,1] (f16)
    __shared__ __align__(16) float    wS[BPW][NL];   // pre-normalization w = s*exp(enc)
    __shared__ __align__(16) _Float16 uS[BPW][NL];   // normalized state (WMMA A input)
    __shared__ __align__(16) float scr[BPW][8];      // finale scratch
    __shared__ float estopS[NL];                     // exp(T[j][STOP])
    __shared__ int   lenS[BPW];

    const int tid  = threadIdx.x;
    const int nt   = tid >> 5;      // wave id = output column tile 0..3
    const int lane = tid & 31;
    const int hl   = lane & 15;
    const int hi   = lane >> 4;
    const int gb0  = blockIdx.x * BPW;

    const int b   = tid >> 3;       // 0..15: batch row for normalize stage
    const int sub = tid & 7;        // 8 threads per batch row

    // ---- preload B fragments of E = exp(transition), kt = 0,1, column block nt ----
    // B k order used here (contiguous, same as A): lanes 0-15: N=lane, K=0..15;
    //                                    lanes 16-31: N=lane-16, K=16..31.
    v16h bf0, bf1;
    {
        const int j = nt * 16 + hl;
        #pragma unroll
        for (int e = 0; e < 16; ++e) {
            const int k0 = hi * 16 + e;        // kt=0
            const int k1 = 32 + hi * 16 + e;   // kt=1
            bf0[e] = (_Float16)__expf(trans[k0 * NL + j]);
            bf1[e] = (_Float16)__expf(trans[k1 * NL + j]);
        }
    }

    // per-lane enc base pointers for the 8 C rows this lane produces
    const int jc = nt * 16 + hl;
    const float* encp[8];
    #pragma unroll
    for (int v = 0; v < 8; ++v)
        encp[v] = enc + ((size_t)(gb0 + v + 8 * hi) * NS) * NL + jc;

    // ---- init: w0[b][i] = exp(T[START][i] + enc[b][0][i]) ----
    if (sub == 0) lenS[b] = lens[gb0 + b];
    if (tid < NL) estopS[tid] = __expf(trans[tid * NL + TSTOP]);
    #pragma unroll
    for (int k = 0; k < 8; ++k) {
        const int i = sub * 8 + k;
        wS[b][i] = __builtin_amdgcn_exp2f(
            (trans[TSTART * NL + i] + enc[((size_t)(gb0 + b) * NS) * NL + i]) * INVLN2);
    }
    __syncthreads();

    const int mylenm1 = lenS[b] - 1;
    float Cb = 0.f;                 // log2 scale, meaningful in sub==0 threads

    // ---- normalize stage for t = 0 ----
    {
        const float4 w0 = *(const float4*)&wS[b][sub * 8];
        const float4 w1 = *(const float4*)&wS[b][sub * 8 + 4];
        float pm = fmaxf(fmaxf(fmaxf(w0.x, w0.y), fmaxf(w0.z, w0.w)),
                         fmaxf(fmaxf(w1.x, w1.y), fmaxf(w1.z, w1.w)));
        pm = group8_max(pm);
        const float r = __builtin_amdgcn_rcpf(pm);
        if (sub == 0) Cb += __builtin_amdgcn_logf(pm);
        v8h up;
        up[0] = (_Float16)(w0.x * r); up[1] = (_Float16)(w0.y * r);
        up[2] = (_Float16)(w0.z * r); up[3] = (_Float16)(w0.w * r);
        up[4] = (_Float16)(w1.x * r); up[5] = (_Float16)(w1.y * r);
        up[6] = (_Float16)(w1.z * r); up[7] = (_Float16)(w1.w * r);
        *(v8h*)&uS[b][sub * 8] = up;
    }
    __syncthreads();

    for (int t = 1; t < NS; ++t) {
        // prefetch enc[.,t,.] and compute ee = exp(enc) OFF the critical path
        float ee[8];
        #pragma unroll
        for (int v = 0; v < 8; ++v)
            ee[v] = __builtin_amdgcn_exp2f(encp[v][(size_t)t * NL] * INVLN2);

        // stage W: WMMA  s(16x16 tile) = U(16x64) * E(64x64), column block nt
        // A k order used here (contiguous, same as B): lanes 0-15: M=lane, K=0..15;
        //                           lanes 16-31: M=lane-16, K=16..31.
        v8h a00 = *(const v8h*)&uS[hl][hi * 16];
        v8h a01 = *(const v8h*)&uS[hl][hi * 16 + 8];
        v8h a10 = *(const v8h*)&uS[hl][32 + hi * 16];
        v8h a11 = *(const v8h*)&uS[hl][32 + hi * 16 + 8];
        v16h a0 = __builtin_shufflevector(a00, a01, 0,1,2,3,4,5,6,7,8,9,10,11,12,13,14,15);
        v16h a1 = __builtin_shufflevector(a10, a11, 0,1,2,3,4,5,6,7,8,9,10,11,12,13,14,15);
        v8f c = {};
        c = __builtin_amdgcn_wmma_f32_16x16x32_f16(false, a0, false, bf0, (short)0, c, false, false);
        c = __builtin_amdgcn_wmma_f32_16x16x32_f16(false, a1, false, bf1, (short)0, c, false, false);

        // w = s * exp(enc)   (C layout: VGPR v -> M = v + 8*hi, N = hl)
        #pragma unroll
        for (int v = 0; v < 8; ++v)
            wS[v + 8 * hi][jc] = c[v] * ee[v];
        __syncthreads();

        // stage N: per-batch linear renormalization (no per-element log/exp)
        const float4 w0 = *(const float4*)&wS[b][sub * 8];
        const float4 w1 = *(const float4*)&wS[b][sub * 8 + 4];
        float pm = fmaxf(fmaxf(fmaxf(w0.x, w0.y), fmaxf(w0.z, w0.w)),
                         fmaxf(fmaxf(w1.x, w1.y), fmaxf(w1.z, w1.w)));
        pm = group8_max(pm);
        if (t <= mylenm1) {                 // freeze row past len-1
            const float r = __builtin_amdgcn_rcpf(pm);
            if (sub == 0) Cb += __builtin_amdgcn_logf(pm);
            v8h up;
            up[0] = (_Float16)(w0.x * r); up[1] = (_Float16)(w0.y * r);
            up[2] = (_Float16)(w0.z * r); up[3] = (_Float16)(w0.w * r);
            up[4] = (_Float16)(w1.x * r); up[5] = (_Float16)(w1.y * r);
            up[6] = (_Float16)(w1.z * r); up[7] = (_Float16)(w1.w * r);
            *(v8h*)&uS[b][sub * 8] = up;
        }
        __syncthreads();
    }

    // ---- unlabeled[b] = ln2 * (Cb + log2( sum_j u[b][j]*exp(T[j][STOP]) )) ----
    if (sub == 0) {
        float dot = 0.f;
        for (int j = 0; j < NL; ++j)
            dot += (float)uS[b][j] * estopS[j];
        out[gb0 + b] = (Cb + __builtin_amdgcn_logf(dot)) * LN2;
    }

    // ---- labeled score: begin + end + sum_t (T[tag_{t-1},tag_t] + enc[b,t,tag_t]) ----
    {
        const int gb  = gb0 + b;
        const int len = mylenm1 + 1;
        float s = 0.f;
        for (int t = 1 + sub; t < NS; t += 8) {
            if (t < len) {
                const int tp = tags[gb * NS + t - 1];
                const int tc = tags[gb * NS + t];
                s += trans[tp * NL + tc] + enc[((size_t)gb * NS + t) * NL + tc];
            }
        }
        scr[b][sub] = s;
        __syncthreads();
        if (sub == 0) {
            float tot = 0.f;
            #pragma unroll
            for (int k = 0; k < 8; ++k) tot += scr[b][k];
            const int t0 = tags[gb * NS];
            const float beginv = trans[TSTART * NL + t0] +
                                 enc[((size_t)gb * NS) * NL + t0];
            const int te = tags[gb * NS + len - 1];
            out[NB + gb] = beginv + trans[te * NL + TSTOP] + tot;
        }
    }
}

extern "C" void kernel_launch(void* const* d_in, const int* in_sizes, int n_in,
                              void* d_out, int out_size, void* d_ws, size_t ws_size,
                              hipStream_t stream) {
    (void)in_sizes; (void)n_in; (void)d_ws; (void)ws_size; (void)out_size;
    const float* enc   = (const float*)d_in[0];
    const float* trans = (const float*)d_in[1];
    const int*   lens  = (const int*)d_in[2];
    const int*   tags  = (const int*)d_in[3];
    float*       out   = (float*)d_out;
    crf_fwd_kernel<<<NB / BPW, 128, 0, stream>>>(enc, trans, lens, tags, out);
}
